// CausalSelfAttention_56332791054959
// MI455X (gfx1250) — compile-verified, matrix operand order checked by reading, not run on hardware
//
#include <hip/hip_runtime.h>
#include <hip/hip_bf16.h>

// ---------------- types ----------------
typedef __attribute__((ext_vector_type(16))) __bf16 bf16x16;
typedef __attribute__((ext_vector_type(8)))  __bf16 bf16x8;
typedef __attribute__((ext_vector_type(8)))  float  floatx8;

#define DEV __device__ __forceinline__

// Load a 16x32 bf16 A/B WMMA fragment from a row-major matrix.
// p points at the tile origin (row0, col0); ld = row stride in elements.
// Per CDNA5 ISA 7.12.2 (16-bit A 16x32): lane<16 -> row=lane, K {0..7,16..23};
// lane>=16 -> row=lane-16, K {8..15,24..31}.  B (KxN) uses the same gather
// from an [N][K] row-major buffer (column n per lane, K packed in VGPRs).
DEV bf16x16 load_frag(const __bf16* __restrict__ p, int ld, int lane) {
    int l = lane & 15;
    int h = lane >> 4;
    const __bf16* q = p + (size_t)l * ld + h * 8;
    bf16x8 lo = *(const bf16x8*)(q);
    bf16x8 hi = *(const bf16x8*)(q + 16);
    return __builtin_shufflevector(lo, hi, 0, 1, 2, 3, 4, 5, 6, 7,
                                           8, 9, 10, 11, 12, 13, 14, 15);
}

DEV floatx8 wmma_bf16(bf16x16 a, bf16x16 b, floatx8 c) {
    return __builtin_amdgcn_wmma_f32_16x16x32_bf16(
        /*neg_a=*/false, a, /*neg_b=*/false, b,
        /*c_mod=*/(short)0, c, /*reuse_a=*/false, /*reuse_b=*/false);
}

// ---------------- fp32 -> bf16 convert ----------------
__global__ __launch_bounds__(256) void cvt_f32_to_bf16(
    const float* __restrict__ in, __bf16* __restrict__ out, int n) {
    int i = blockIdx.x * blockDim.x + threadIdx.x;
    if (i < n) out[i] = (__bf16)in[i];
}

// ---------------- generic GEMM: C[M,N] = A[M,K] @ W[N,K]^T ----------------
// A, W bf16 row-major; C fp32 row-major.
// One wave -> 32(M) x 64(N) tile: 2 A-frags x 4 B-frags = 8 WMMAs / 32-K step.
// K-loop is software-pipelined: next iteration's fragments are loaded before
// the current iteration's WMMAs so loads run a full WMMA bundle ahead.
__global__ __launch_bounds__(128) void gemm_xwt_bf16(
    const __bf16* __restrict__ A, const __bf16* __restrict__ W,
    float* __restrict__ C, int M, int N, int K) {
    int lane = threadIdx.x & 31;
    int wid  = blockIdx.x * (blockDim.x >> 5) + (threadIdx.x >> 5);
    int tiles_n = N >> 6;
    int tm = wid / tiles_n;
    int tn = wid % tiles_n;
    if (tm * 32 >= M) return;
    int m0 = tm * 32, n0 = tn * 64;

    const __bf16* A0 = A + (size_t)m0 * K;
    const __bf16* A1 = A + (size_t)(m0 + 16) * K;
    const __bf16* Wp = W + (size_t)n0 * K;

    floatx8 acc[2][4] = {};

    // prologue: fragments for k0 = 0
    bf16x16 a0 = load_frag(A0, K, lane);
    bf16x16 a1 = load_frag(A1, K, lane);
    bf16x16 b0 = load_frag(Wp,                     K, lane);
    bf16x16 b1 = load_frag(Wp + (size_t)16 * K,    K, lane);
    bf16x16 b2 = load_frag(Wp + (size_t)32 * K,    K, lane);
    bf16x16 b3 = load_frag(Wp + (size_t)48 * K,    K, lane);

    for (int k0 = 32; k0 < K; k0 += 32) {
        // prefetch next K-step fragments (issues loads before the WMMAs)
        bf16x16 na0 = load_frag(A0 + k0, K, lane);
        bf16x16 na1 = load_frag(A1 + k0, K, lane);
        bf16x16 nb0 = load_frag(Wp + k0,                  K, lane);
        bf16x16 nb1 = load_frag(Wp + (size_t)16 * K + k0, K, lane);
        bf16x16 nb2 = load_frag(Wp + (size_t)32 * K + k0, K, lane);
        bf16x16 nb3 = load_frag(Wp + (size_t)48 * K + k0, K, lane);

        acc[0][0] = wmma_bf16(a0, b0, acc[0][0]);
        acc[1][0] = wmma_bf16(a1, b0, acc[1][0]);
        acc[0][1] = wmma_bf16(a0, b1, acc[0][1]);
        acc[1][1] = wmma_bf16(a1, b1, acc[1][1]);
        acc[0][2] = wmma_bf16(a0, b2, acc[0][2]);
        acc[1][2] = wmma_bf16(a1, b2, acc[1][2]);
        acc[0][3] = wmma_bf16(a0, b3, acc[0][3]);
        acc[1][3] = wmma_bf16(a1, b3, acc[1][3]);

        a0 = na0; a1 = na1;
        b0 = nb0; b1 = nb1; b2 = nb2; b3 = nb3;
    }
    // epilogue K-step
    acc[0][0] = wmma_bf16(a0, b0, acc[0][0]);
    acc[1][0] = wmma_bf16(a1, b0, acc[1][0]);
    acc[0][1] = wmma_bf16(a0, b1, acc[0][1]);
    acc[1][1] = wmma_bf16(a1, b1, acc[1][1]);
    acc[0][2] = wmma_bf16(a0, b2, acc[0][2]);
    acc[1][2] = wmma_bf16(a1, b2, acc[1][2]);
    acc[0][3] = wmma_bf16(a0, b3, acc[0][3]);
    acc[1][3] = wmma_bf16(a1, b3, acc[1][3]);

    int cl = lane & 15, hf = lane >> 4;
#pragma unroll
    for (int i = 0; i < 2; ++i)
#pragma unroll
        for (int j = 0; j < 4; ++j)
#pragma unroll
            for (int r = 0; r < 8; ++r)
                C[(size_t)(m0 + 16 * i + r + 8 * hf) * N + (n0 + 16 * j + cl)] =
                    acc[i][j][r];
}

// ---------------- RoPE + rearrange ----------------
// Qt/Kt/Vt_in: fp32 [B,S,D].  Outputs:
//   Qb,Kb bf16 [B*H, S, 64]  (post-RoPE)
//   Vtb   bf16 [B*H, 64, S]  (transposed V for PV B-fragments)
__global__ __launch_bounds__(256) void rope_rearrange(
    const float* __restrict__ Qt, const float* __restrict__ Kt,
    const float* __restrict__ Vt_in,
    const float* __restrict__ fc, const float* __restrict__ fs,
    __bf16* __restrict__ Qb, __bf16* __restrict__ Kb,
    __bf16* __restrict__ Vtb) {
    int idx = blockIdx.x * blockDim.x + threadIdx.x;   // pair index
    if (idx >= 2 * 2048 * 512) return;                 // B*S*(D/2)
    int p  = idx & 511;                                // pair within row
    int bs = idx >> 9;                                 // b*2048 + s
    int s  = bs & 2047;
    int b  = bs >> 11;
    int h  = p >> 5;                                   // head
    int i  = p & 31;                                   // pair within head
    size_t off = (size_t)bs * 1024 + h * 64 + 2 * i;

    float c  = fc[s * 32 + i];
    float sn = fs[s * 32 + i];
    float qr = Qt[off], qi = Qt[off + 1];
    float kr = Kt[off], ki = Kt[off + 1];
    float v0 = Vt_in[off], v1 = Vt_in[off + 1];

    size_t hb = (size_t)(b * 16 + h);
    size_t qk = (hb * 2048 + s) * 64 + 2 * i;
    Qb[qk]     = (__bf16)(qr * c - qi * sn);
    Qb[qk + 1] = (__bf16)(qi * c + qr * sn);
    Kb[qk]     = (__bf16)(kr * c - ki * sn);
    Kb[qk + 1] = (__bf16)(ki * c + kr * sn);
    size_t vo = (hb * 64 + 2 * i) * 2048 + s;
    Vtb[vo]        = (__bf16)v0;
    Vtb[vo + 2048] = (__bf16)v1;
}

// ---------------- flash-style causal attention ----------------
// One wave handles 16 query rows of one (b,h).  Keys in blocks of 32.
// Q,K bf16 [B*H,S,64]; Vt bf16 [B*H,64,S]; O bf16 [B,S,D] (attn_cat).
__global__ __launch_bounds__(128) void flash_attn_causal(
    const __bf16* __restrict__ Q, const __bf16* __restrict__ Kv,
    const __bf16* __restrict__ Vt, __bf16* __restrict__ O) {
    __shared__ __bf16 lds[4][16 * 32];     // per-wave P tile (16x32 bf16)
    int lane = threadIdx.x & 31;
    int wv   = threadIdx.x >> 5;
    int wid  = blockIdx.x * 4 + wv;        // 0..4095
    int qt   = wid & 127;
    int bh   = wid >> 7;                   // 0..31
    int q0   = qt << 4;
    int cl = lane & 15, hf = lane >> 4;

    const __bf16* Qh = Q  + (size_t)bh * 2048 * 64;
    const __bf16* Kh = Kv + (size_t)bh * 2048 * 64;
    const __bf16* Vh = Vt + (size_t)bh * 64 * 2048;

    bf16x16 aq0 = load_frag(Qh + (size_t)q0 * 64,      64, lane);
    bf16x16 aq1 = load_frag(Qh + (size_t)q0 * 64 + 32, 64, lane);

    floatx8 acc[4] = {};
    float m_run[8], l_run[8];
#pragma unroll
    for (int r = 0; r < 8; ++r) { m_run[r] = -1e30f; l_run[r] = 0.0f; }

    __bf16* plds = &lds[wv][0];
    int nkb = (q0 + 16 + 31) >> 5;         // key blocks covering rows <= q0+15
    for (int kb = 0; kb < nkb; ++kb) {
        int kbase = kb << 5;
        // S-tiles: scores[16 x 32] = Q(16x64) @ K(32x64)^T
        floatx8 s0 = {}, s1 = {};
        s0 = wmma_bf16(aq0, load_frag(Kh + (size_t)kbase * 64,             64, lane), s0);
        s0 = wmma_bf16(aq1, load_frag(Kh + (size_t)kbase * 64 + 32,        64, lane), s0);
        s1 = wmma_bf16(aq0, load_frag(Kh + (size_t)(kbase + 16) * 64,      64, lane), s1);
        s1 = wmma_bf16(aq1, load_frag(Kh + (size_t)(kbase + 16) * 64 + 32, 64, lane), s1);

        // scale + causal mask + per-row max over this block
        float t[8];
#pragma unroll
        for (int r = 0; r < 8; ++r) {
            int q = q0 + r + 8 * hf;
            float e0 = (kbase + cl      <= q) ? s0[r] * 0.125f : -1e30f;
            float e1 = (kbase + 16 + cl <= q) ? s1[r] * 0.125f : -1e30f;
            s0[r] = e0; s1[r] = e1;
            t[r] = fmaxf(e0, e1);
        }
#pragma unroll
        for (int off = 1; off < 16; off <<= 1)
#pragma unroll
            for (int r = 0; r < 8; ++r)
                t[r] = fmaxf(t[r], __shfl_xor(t[r], off, 32));

        // online softmax update
        float p0[8], p1[8], rs[8];
#pragma unroll
        for (int r = 0; r < 8; ++r) {
            float mn = fmaxf(m_run[r], t[r]);
            float alpha = __expf(m_run[r] - mn);
            m_run[r] = mn;
            l_run[r] *= alpha;
#pragma unroll
            for (int j = 0; j < 4; ++j) acc[j][r] *= alpha;
            p0[r] = __expf(s0[r] - mn);
            p1[r] = __expf(s1[r] - mn);
            rs[r] = p0[r] + p1[r];
        }
#pragma unroll
        for (int off = 1; off < 16; off <<= 1)
#pragma unroll
            for (int r = 0; r < 8; ++r)
                rs[r] += __shfl_xor(rs[r], off, 32);
#pragma unroll
        for (int r = 0; r < 8; ++r) l_run[r] += rs[r];

        // C-layout -> row-major P tile in LDS (16x32 bf16), then A-fragment
#pragma unroll
        for (int r = 0; r < 8; ++r) {
            int row = r + 8 * hf;
            plds[row * 32 + cl]      = (__bf16)p0[r];
            plds[row * 32 + cl + 16] = (__bf16)p1[r];
        }
        bf16x16 pa = load_frag(plds, 32, lane);   // same-wave LDS: in-order

        // acc += P(16x32) @ V(32x64)  (B-fragments from transposed V)
#pragma unroll
        for (int j = 0; j < 4; ++j) {
            bf16x16 bv = load_frag(Vh + (size_t)(16 * j) * 2048 + kbase, 2048, lane);
            acc[j] = wmma_bf16(pa, bv, acc[j]);
        }
    }

    // normalize + store attn_cat[b][q][h*64 + d] as bf16
    int b = bh >> 4, h = bh & 15;
#pragma unroll
    for (int r = 0; r < 8; ++r) {
        float inv = 1.0f / l_run[r];
        int q = q0 + r + 8 * hf;
        __bf16* orow = O + ((size_t)(b * 2048 + q)) * 1024 + h * 64;
#pragma unroll
        for (int j = 0; j < 4; ++j)
            orow[16 * j + cl] = (__bf16)(acc[j][r] * inv);
    }
}

// ---------------- host launcher ----------------
extern "C" void kernel_launch(void* const* d_in, const int* in_sizes, int n_in,
                              void* d_out, int out_size, void* d_ws, size_t ws_size,
                              hipStream_t stream) {
    (void)in_sizes; (void)n_in; (void)out_size; (void)ws_size;
    const int B = 2, S = 2048, D = 1024;       // H=16, hd=64
    const int M = B * S;                       // 4096 rows

    const float* x  = (const float*)d_in[0];
    const float* fc = (const float*)d_in[1];
    const float* fs = (const float*)d_in[2];
    const float* Wq = (const float*)d_in[3];
    const float* Wk = (const float*)d_in[4];
    const float* Wv = (const float*)d_in[5];
    const float* Wo = (const float*)d_in[6];
    float* out = (float*)d_out;

    // workspace carve-up
    char* ws = (char*)d_ws;
    size_t o = 0;
    __bf16* xb  = (__bf16*)(ws + o); o += (size_t)M * D * 2;       // 8 MiB
    __bf16* wqb = (__bf16*)(ws + o); o += (size_t)D * D * 2;       // 2 MiB
    __bf16* wkb = (__bf16*)(ws + o); o += (size_t)D * D * 2;
    __bf16* wvb = (__bf16*)(ws + o); o += (size_t)D * D * 2;
    __bf16* wob = (__bf16*)(ws + o); o += (size_t)D * D * 2;
    float* Qtmp = (float*)(ws + o);  o += (size_t)M * D * 4;       // 16 MiB
    float* Ktmp = (float*)(ws + o);  o += (size_t)M * D * 4;
    float* Vtmp = (float*)(ws + o);  o += (size_t)M * D * 4;
    __bf16* Qb  = (__bf16*)(ws + o); o += (size_t)M * D * 2;       // [B*H,S,64]
    __bf16* Kb  = (__bf16*)(ws + o); o += (size_t)M * D * 2;
    __bf16* Vtb = (__bf16*)(ws + o); o += (size_t)M * D * 2;       // [B*H,64,S]
    __bf16* attnb = (__bf16*)(ws + o); o += (size_t)M * D * 2;     // [B,S,D]

    // 1) convert inputs to bf16
    {
        int n = M * D;
        cvt_f32_to_bf16<<<(n + 255) / 256, 256, 0, stream>>>(x, xb, n);
        n = D * D;
        cvt_f32_to_bf16<<<(n + 255) / 256, 256, 0, stream>>>(Wq, wqb, n);
        cvt_f32_to_bf16<<<(n + 255) / 256, 256, 0, stream>>>(Wk, wkb, n);
        cvt_f32_to_bf16<<<(n + 255) / 256, 256, 0, stream>>>(Wv, wvb, n);
        cvt_f32_to_bf16<<<(n + 255) / 256, 256, 0, stream>>>(Wo, wob, n);
    }

    // 2) QKV projections: waves = (M/32)*(D/64) = 2048 -> 512 blocks x 128
    int gemm_blocks = (M / 32) * (D / 64) / 4;
    gemm_xwt_bf16<<<gemm_blocks, 128, 0, stream>>>(xb, wqb, Qtmp, M, D, D);
    gemm_xwt_bf16<<<gemm_blocks, 128, 0, stream>>>(xb, wkb, Ktmp, M, D, D);
    gemm_xwt_bf16<<<gemm_blocks, 128, 0, stream>>>(xb, wvb, Vtmp, M, D, D);

    // 3) RoPE + rearrange to per-head layouts
    {
        int n = B * S * (D / 2);                 // 2,097,152 pair threads
        rope_rearrange<<<(n + 255) / 256, 256, 0, stream>>>(
            Qtmp, Ktmp, Vtmp, fc, fs, Qb, Kb, Vtb);
    }

    // 4) causal flash attention: 4096 waves -> 1024 blocks x 128
    flash_attn_causal<<<1024, 128, 0, stream>>>(Qb, Kb, Vtb, attnb);

    // 5) output projection -> fp32 result
    gemm_xwt_bf16<<<gemm_blocks, 128, 0, stream>>>(attnb, wob, out, M, D, D);
}
